// Model_65395172049168
// MI455X (gfx1250) — matrix operand order repaired and checked by reading, not compiled, not run on hardware
//
#include <hip/hip_runtime.h>
#include <hip/hip_bf16.h>

// ---------------------------------------------------------------------------
// Types for CDNA5 WMMA (wave32)
// ---------------------------------------------------------------------------
typedef __attribute__((ext_vector_type(16))) __bf16 v16bf;
typedef __attribute__((ext_vector_type(8)))  __bf16 v8bf;
typedef __attribute__((ext_vector_type(8)))  float  v8f;

#define NXR      32768
#define JR       1024
#define DIMR     28
#define TOTROWS  66560      // 2*NX + J
#define FSTRIDE  112        // padded f width (100 -> 112)
#define GSTRIDE  32         // padded g width (28 -> 32)
#define ACTW     320        // padded hidden width for K (300 -> 320)

// ---------------------------------------------------------------------------
// Fragment builders.
// A (16-bit 16x32): lane m=l%16; elems j: k = kt*32 + (l<16?0:16) + j
//   -> two contiguous 16B loads per lane.
// B (16-bit 32x16): lane n=l%16; elems j: k = kt*32 + (l<16?0:16) + j
//   -> one contiguous 32B load per lane.
// ---------------------------------------------------------------------------
__device__ __forceinline__ v16bf frag_a(const __bf16* rowbase, int kt, int lane) {
  int kb = kt * 32 + ((lane & 16) ? 16 : 0);
  v8bf lo = *(const v8bf*)(rowbase + kb);
  v8bf hi = *(const v8bf*)(rowbase + kb + 8);
  v16bf r;
#pragma unroll
  for (int i = 0; i < 8; ++i) { r[i] = lo[i]; r[i + 8] = hi[i]; }
  return r;
}

__device__ __forceinline__ v16bf frag_b_w(const __bf16* tile, int lane) {
  const v8bf* p = (const v8bf*)(tile + lane * 16);
  v8bf lo = p[0], hi = p[1];
  v16bf r;
#pragma unroll
  for (int i = 0; i < 8; ++i) { r[i] = lo[i]; r[i + 8] = hi[i]; }
  return r;
}

__device__ __forceinline__ v16bf frag_b_feat(const __bf16* base, int stride,
                                             int vrow0, int kt, int lane) {
  const __bf16* p = base + (size_t)(vrow0 + (lane & 15)) * stride
                    + kt * 32 + ((lane & 16) ? 16 : 0);
  v8bf lo = *(const v8bf*)(p);
  v8bf hi = *(const v8bf*)(p + 8);
  v16bf r;
#pragma unroll
  for (int i = 0; i < 8; ++i) { r[i] = lo[i]; r[i + 8] = hi[i]; }
  return r;
}

__device__ __forceinline__ v8f wmma_bf16(v16bf a, v16bf b, v8f c) {
  return __builtin_amdgcn_wmma_f32_16x16x32_bf16(false, a, false, b,
                                                 (short)0, c, false, false);
}

// ---------------------------------------------------------------------------
// Weight conversion: fp32 row-major [K][N] -> bf16 fragment-swizzled tiles.
// One block (512 threads) per 32x16 tile; thread t -> lane l=t/16, elem j=t%16.
// ---------------------------------------------------------------------------
__global__ __launch_bounds__(512)
void conv_weights_kernel(const float* __restrict__ W, __bf16* __restrict__ dst,
                         int K, int N, int KT) {
  int tile = blockIdx.x;
  int nt = tile / KT, kt = tile % KT;
  int t = threadIdx.x;
  int l = t >> 4, j = t & 15;
  int n = nt * 16 + (l & 15);
  int k = kt * 32 + ((l & 16) ? 16 : 0) + j;
  float v = (k < K && n < N) ? W[(size_t)k * N + n] : 0.0f;
  dst[(size_t)tile * 512 + l * 16 + j] = (__bf16)v;
}

__global__ void init_zero_kernel(float* colsum, float* scalars) {
  int t = threadIdx.x + blockIdx.x * blockDim.x;
  for (int i = t; i < 1024; i += blockDim.x * gridDim.x) colsum[i] = 0.0f;
  if (t < 4) scalars[t] = 0.0f;
}

// ---------------------------------------------------------------------------
// Feature kernel: runs both MLPs for a block of 128 rows.
// 256 threads = 8 waves; wave w owns rows [w*16, w*16+16).
// ---------------------------------------------------------------------------
struct K1Params {
  const float* XY;
  const float* V;
  const float* bias[12];     // dn 0..5, an 6..11
  const __bf16* wsw[12];     // swizzled bf16 weights
  __bf16* f;                 // [TOTROWS][FSTRIDE]
  __bf16* g;                 // [TOTROWS][GSTRIDE]
  float*  fn;                // [TOTROWS]
  float*  gn;                // [TOTROWS]
};

__device__ __forceinline__ void hidden_layer(const __bf16* in, int inStride, int KT,
                                             __bf16* out, const __bf16* wsw,
                                             const float* bias, int N, int NT,
                                             int wave, int lane) {
  const __bf16* arow = in + (size_t)(wave * 16 + (lane & 15)) * inStride;
  int rowoff = wave * 16 + ((lane & 16) ? 8 : 0);
  for (int nt = 0; nt < NT; ++nt) {
    v8f acc = {};
    for (int kt = 0; kt < KT; ++kt) {
      v16bf a = frag_a(arow, kt, lane);
      v16bf b = frag_b_w(wsw + (size_t)(nt * KT + kt) * 512, lane);
      acc = wmma_bf16(a, b, acc);
    }
    int n = nt * 16 + (lane & 15);
    float bv = (n < N) ? bias[n] : 0.0f;
#pragma unroll
    for (int r = 0; r < 8; ++r) {
      float v = acc[r] + bv;
      v = v > 0.0f ? v : 0.0f;          // ReLU
      out[(size_t)(rowoff + r) * ACTW + n] = (__bf16)v;
    }
  }
}

__global__ __launch_bounds__(256)
void features_kernel(K1Params p) {
  extern __shared__ char smem[];
  __bf16* buf0 = (__bf16*)smem;                 // 128*320 bf16
  __bf16* buf1 = buf0 + 128 * ACTW;             // 128*320 bf16
  __bf16* inb  = buf1 + 128 * ACTW;             // 128*32  bf16
  float*  inf  = (float*)(inb + 128 * 32);      // 128*28  f32
  float*  fns  = inf + 128 * 28;                // 128 f32
  float*  gns  = fns + 128;                     // 128 f32

  int tid = threadIdx.x, wave = tid >> 5, lane = tid & 31;
  long rb = (long)blockIdx.x * 128;

  // Stage inputs (fp32 copy for residual, bf16 zero-padded copy for GEMM)
  for (int i = tid; i < 128 * DIMR; i += 256) {
    int r = i / DIMR, c = i % DIMR;
    long row = rb + r;
    float v = (row < 2 * NXR) ? p.XY[row * DIMR + c]
                              : p.V[(row - 2 * NXR) * DIMR + c];
    inf[r * DIMR + c] = v;
    inb[r * 32 + c] = (__bf16)v;
  }
  for (int i = tid; i < 128 * 4; i += 256) {
    int r = i / 4, c = 28 + (i & 3);
    inb[r * 32 + c] = (__bf16)0.0f;
  }
  // Zero pad columns 304..319 of both activation buffers (never rewritten)
  for (int i = tid; i < 128 * 16; i += 256) {
    int r = i / 16, c = 304 + (i % 16);
    buf0[(size_t)r * ACTW + c] = (__bf16)0.0f;
    buf1[(size_t)r * ACTW + c] = (__bf16)0.0f;
  }
  for (int i = tid; i < 128; i += 256) { fns[i] = 0.0f; gns[i] = 0.0f; }
  __syncthreads();

  for (int net = 0; net < 2; ++net) {
    const int base = net * 6;
    // Layer 0: 28(->32) -> 300, KT=1, NT=19
    hidden_layer(inb, 32, 1, buf0, p.wsw[base + 0], p.bias[base + 0], 300, 19, wave, lane);
    __syncthreads();
    // Layers 1..4: 300(->320) -> 300, KT=10, NT=19, ping-pong
    __bf16* a = buf0; __bf16* b = buf1;
    for (int l = 1; l < 5; ++l) {
      hidden_layer(a, ACTW, 10, b, p.wsw[base + l], p.bias[base + l], 300, 19, wave, lane);
      __syncthreads();
      __bf16* t2 = a; a = b; b = t2;
    }
    // final layer input is now in `a` (== buf0 after 4 swaps)
    const __bf16* arow = a + (size_t)(wave * 16 + (lane & 15)) * ACTW;
    int rowoff = wave * 16 + ((lane & 16) ? 8 : 0);
    if (net == 0) {
      // dn L5: 300 -> 100 (pad 112), no ReLU; write f + fnorm
      for (int nt = 0; nt < 7; ++nt) {
        v8f acc = {};
        for (int kt = 0; kt < 10; ++kt)
          acc = wmma_bf16(frag_a(arow, kt, lane),
                          frag_b_w(p.wsw[5] + (size_t)(nt * 10 + kt) * 512, lane), acc);
        int n = nt * 16 + (lane & 15);
        float bv = (n < 100) ? p.bias[5][n] : 0.0f;
#pragma unroll
        for (int r = 0; r < 8; ++r) {
          float v = acc[r] + bv;
          long grow = rb + rowoff + r;
          p.f[grow * FSTRIDE + n] = (__bf16)v;
          if (n < 100) atomicAdd(&fns[rowoff + r], v * v);
        }
      }
    } else {
      // an L5: 300 -> 28 (pad 32), no ReLU, + residual x; write g + gnorm
      for (int nt = 0; nt < 2; ++nt) {
        v8f acc = {};
        for (int kt = 0; kt < 10; ++kt)
          acc = wmma_bf16(frag_a(arow, kt, lane),
                          frag_b_w(p.wsw[11] + (size_t)(nt * 10 + kt) * 512, lane), acc);
        int n = nt * 16 + (lane & 15);
        float bv = (n < DIMR) ? p.bias[11][n] : 0.0f;
#pragma unroll
        for (int r = 0; r < 8; ++r) {
          float v = acc[r] + bv;
          if (n < DIMR) v += inf[(rowoff + r) * DIMR + n];   // residual
          long grow = rb + rowoff + r;
          p.g[grow * GSTRIDE + n] = (__bf16)v;
          if (n < DIMR) atomicAdd(&gns[rowoff + r], v * v);
        }
      }
    }
    __syncthreads();
  }
  for (int i = tid; i < 128; i += 256) {
    p.fn[rb + i] = fns[i];
    p.gn[rb + i] = gns[i];
  }
}

// ---------------------------------------------------------------------------
// UME pass: 512 WGs x 64 X-rows (+ paired Y-rows). 8 waves: wave w handles
// M-subtile (w&3) and column-tile parity (w>>2) over 64 column tiles (J=1024).
// pass==1: accumulate column sums (-> mu) and sum(fm^2).
// pass==2: accumulate sum((fm @ mu)^2).
// ---------------------------------------------------------------------------
__global__ __launch_bounds__(256)
void ume_pass_kernel(const __bf16* __restrict__ f, const __bf16* __restrict__ g,
                     const float* __restrict__ fn, const float* __restrict__ gn,
                     const float* __restrict__ epsP, const float* __restrict__ sigP,
                     const float* __restrict__ sig0P, const float* __restrict__ cstP,
                     float* __restrict__ colsum_g, float* __restrict__ sumfm2_g,
                     const float* __restrict__ mu, float* __restrict__ sumS2_g,
                     int pass) {
  __shared__ float cs[1024];
  __shared__ float rowdot[64];
  int tid = threadIdx.x, wave = tid >> 5, lane = tid & 31;
  int sub = wave & 3, par = wave >> 2;
  long xb = (long)blockIdx.x * 64;

  if (pass == 1) { for (int i = tid; i < 1024; i += 256) cs[i] = 0.0f; }
  else           { for (int i = tid; i < 64;   i += 256) rowdot[i] = 0.0f; }
  __syncthreads();

  float ep = 1.0f / (1.0f + __expf(-epsP[0]));
  float inv_s0 = 1.0f / (sig0P[0] * sig0P[0]);
  float inv_s  = 1.0f / (sigP[0]  * sigP[0]);
  float cst = cstP[0];
  const float invSqrtJ = 0.03125f;   // 1/sqrt(1024)

  long mrowX = xb + sub * 16 + (lane & 15);
  const __bf16* faX = f + mrowX * FSTRIDE;
  const __bf16* faY = f + (mrowX + NXR) * FSTRIDE;
  const __bf16* gaX = g + mrowX * GSTRIDE;
  const __bf16* gaY = g + (mrowX + NXR) * GSTRIDE;

  int rbase = sub * 16 + ((lane & 16) ? 8 : 0);
  float fnX[8], fnY[8], gnX[8], gnY[8];
#pragma unroll
  for (int r = 0; r < 8; ++r) {
    long rowX = xb + rbase + r;
    fnX[r] = fn[rowX];        gnX[r] = gn[rowX];
    fnY[r] = fn[rowX + NXR];  gnY[r] = gn[rowX + NXR];
  }

  float sumsq = 0.0f;
  float rd[8];
#pragma unroll
  for (int r = 0; r < 8; ++r) rd[r] = 0.0f;

  for (int nt = par; nt < 64; nt += 2) {
    v8f dfX = {}, dfY = {}, dgX = {}, dgY = {};
    for (int kt = 0; kt < 7; ++kt) {
      v16bf bv = frag_b_feat(f, FSTRIDE, 2 * NXR + nt * 16, kt, lane);
      dfX = wmma_bf16(frag_a(faX, kt, lane), bv, dfX);
      dfY = wmma_bf16(frag_a(faY, kt, lane), bv, dfY);
    }
    {
      v16bf bg = frag_b_feat(g, GSTRIDE, 2 * NXR + nt * 16, 0, lane);
      dgX = wmma_bf16(frag_a(gaX, 0, lane), bg, dgX);
      dgY = wmma_bf16(frag_a(gaY, 0, lane), bg, dgY);
    }
    int n = nt * 16 + (lane & 15);
    float fnv = fn[2 * NXR + n];
    float gnv = gn[2 * NXR + n];
    float mun = (pass == 2) ? mu[n] : 0.0f;
    float colpart = 0.0f;
#pragma unroll
    for (int r = 0; r < 8; ++r) {
      float Dfx = fnX[r] + fnv - 2.0f * dfX[r]; Dfx = fmaxf(Dfx, 0.0f);
      float Dgx = gnX[r] + gnv - 2.0f * dgX[r]; Dgx = fmaxf(Dgx, 0.0f);
      float KX = cst * ((1.0f - ep) * __expf(-Dfx * inv_s0) + ep) * __expf(-Dgx * inv_s);
      float Dfy = fnY[r] + fnv - 2.0f * dfY[r]; Dfy = fmaxf(Dfy, 0.0f);
      float Dgy = gnY[r] + gnv - 2.0f * dgY[r]; Dgy = fmaxf(Dgy, 0.0f);
      float KY = cst * ((1.0f - ep) * __expf(-Dfy * inv_s0) + ep) * __expf(-Dgy * inv_s);
      float fmv = (KX - KY) * invSqrtJ;
      if (pass == 1) { colpart += fmv; sumsq += fmv * fmv; }
      else           { rd[r] += fmv * mun; }
    }
    if (pass == 1) atomicAdd(&cs[n], colpart);
  }
  __syncthreads();

  if (pass == 1) {
#pragma unroll
    for (int off = 16; off >= 1; off >>= 1) sumsq += __shfl_xor(sumsq, off, 32);
    if (lane == 0) atomicAdd(sumfm2_g, sumsq);
    for (int i = tid; i < 1024; i += 256) atomicAdd(&colsum_g[i], cs[i]);
  } else {
#pragma unroll
    for (int r = 0; r < 8; ++r) {
      float v = rd[r];
#pragma unroll
      for (int off = 1; off < 16; off <<= 1) v += __shfl_xor(v, off, 16);
      rd[r] = v;
    }
    if ((lane & 15) == 0) {
#pragma unroll
      for (int r = 0; r < 8; ++r) atomicAdd(&rowdot[rbase + r], rd[r]);
    }
    __syncthreads();
    for (int i = tid; i < 64; i += 256) {
      float s = rowdot[i];
      atomicAdd(sumS2_g, s * s);
    }
  }
}

// ---------------------------------------------------------------------------
// mu finalize + sum(mu^2)
// ---------------------------------------------------------------------------
__global__ __launch_bounds__(256)
void finalize_mu_kernel(const float* __restrict__ colsum, float* __restrict__ mu,
                        float* __restrict__ sum_mu2) {
  __shared__ float red[256];
  int t = threadIdx.x;
  float s = 0.0f;
  for (int i = t; i < 1024; i += 256) {
    float m = colsum[i] * (1.0f / (float)NXR);
    mu[i] = m;
    s += m * m;
  }
  red[t] = s;
  __syncthreads();
  for (int o = 128; o > 0; o >>= 1) {
    if (t < o) red[t] += red[t + o];
    __syncthreads();
  }
  if (t == 0) *sum_mu2 = red[0];
}

__global__ void final_kernel(const float* sum_mu2_p, const float* sumfm2_p,
                             const float* sumS2_p, float* out) {
  float sum_mu2 = *sum_mu2_p, sumfm2 = *sumfm2_p, sumS2 = *sumS2_p;
  const float nx = (float)NXR;
  float t1 = sum_mu2 * (nx / (nx - 1.0f));
  float t2 = sumfm2 / (nx * (nx - 1.0f));
  float mean = t1 - t2;
  float var = 4.0f * (sumS2 / nx) - 4.0f * sum_mu2 * sum_mu2;
  out[0] = -(mean / sqrtf(var + 1e-6f));
}

// ---------------------------------------------------------------------------
// Host launcher
// ---------------------------------------------------------------------------
extern "C" void kernel_launch(void* const* d_in, const int* in_sizes, int n_in,
                              void* d_out, int out_size, void* d_ws, size_t ws_size,
                              hipStream_t stream) {
  (void)in_sizes; (void)n_in; (void)out_size; (void)ws_size;

  // Layer geometry (K, N, KT=ceil(K/32), NT=ceil(N/16)); dn 0..5, an 6..11
  static const int Ks[12]  = {28, 300, 300, 300, 300, 300, 28, 300, 300, 300, 300, 300};
  static const int Ns[12]  = {300, 300, 300, 300, 300, 100, 300, 300, 300, 300, 300, 28};
  static const int KTs[12] = {1, 10, 10, 10, 10, 10, 1, 10, 10, 10, 10, 10};
  static const int NTs[12] = {19, 19, 19, 19, 19, 7, 19, 19, 19, 19, 19, 2};

  // Workspace partition
  char* ws = (char*)d_ws;
  size_t off = 0;
  auto take = [&](size_t bytes) {
    void* p = ws + off;
    off += (bytes + 255) & ~(size_t)255;
    return p;
  };
  __bf16* wsw[12];
  for (int i = 0; i < 12; ++i)
    wsw[i] = (__bf16*)take((size_t)KTs[i] * NTs[i] * 512 * sizeof(__bf16));
  __bf16* fbuf = (__bf16*)take((size_t)TOTROWS * FSTRIDE * sizeof(__bf16));
  __bf16* gbuf = (__bf16*)take((size_t)TOTROWS * GSTRIDE * sizeof(__bf16));
  float*  fnb  = (float*)take((size_t)TOTROWS * sizeof(float));
  float*  gnb  = (float*)take((size_t)TOTROWS * sizeof(float));
  float*  colsum = (float*)take(1024 * sizeof(float));
  float*  mu     = (float*)take(1024 * sizeof(float));
  float*  scalars = (float*)take(4 * sizeof(float));  // [0]=sumfm2 [1]=sumS2 [2]=sum_mu2

  const float* XY = (const float*)d_in[0];
  const float* V  = (const float*)d_in[1];
  const float* Wp[12];
  const float* Bp[12];
  for (int i = 0; i < 6; ++i) {
    Wp[i]     = (const float*)d_in[2 + 2 * i];
    Bp[i]     = (const float*)d_in[3 + 2 * i];
    Wp[6 + i] = (const float*)d_in[14 + 2 * i];
    Bp[6 + i] = (const float*)d_in[15 + 2 * i];
  }
  const float* epsP  = (const float*)d_in[26];
  const float* sigP  = (const float*)d_in[27];
  const float* sig0P = (const float*)d_in[28];
  const float* cstP  = (const float*)d_in[29];

  // 0) zero accumulators
  init_zero_kernel<<<4, 256, 0, stream>>>(colsum, scalars);

  // 1) weight conversion to bf16 swizzled fragments
  for (int i = 0; i < 12; ++i) {
    int tiles = KTs[i] * NTs[i];
    conv_weights_kernel<<<tiles, 512, 0, stream>>>(Wp[i], wsw[i], Ks[i], Ns[i], KTs[i]);
  }

  // 2) features (both MLPs) for all 66560 rows
  K1Params p;
  p.XY = XY; p.V = V;
  for (int i = 0; i < 12; ++i) { p.bias[i] = Bp[i]; p.wsw[i] = wsw[i]; }
  p.f = fbuf; p.g = gbuf; p.fn = fnb; p.gn = gnb;
  size_t smem = (size_t)128 * ACTW * 2 * 2   // buf0+buf1 bf16
              + (size_t)128 * 32 * 2          // inb bf16
              + (size_t)128 * DIMR * 4        // inf f32
              + (size_t)128 * 2 * 4;          // fns+gns
  features_kernel<<<TOTROWS / 128, 256, smem, stream>>>(p);

  // 3) pass 1: column sums + sum(fm^2)
  ume_pass_kernel<<<NXR / 64, 256, 0, stream>>>(fbuf, gbuf, fnb, gnb,
                                                epsP, sigP, sig0P, cstP,
                                                colsum, &scalars[0],
                                                mu, &scalars[1], 1);

  // 4) mu = colsum/nx, sum(mu^2)
  finalize_mu_kernel<<<1, 256, 0, stream>>>(colsum, mu, &scalars[2]);

  // 5) pass 2: sum((fm @ mu)^2)  (fm recomputed, avoids 134MB spill)
  ume_pass_kernel<<<NXR / 64, 256, 0, stream>>>(fbuf, gbuf, fnb, gnb,
                                                epsP, sigP, sig0P, cstP,
                                                colsum, &scalars[0],
                                                mu, &scalars[1], 2);

  // 6) final statistic
  final_kernel<<<1, 1, 0, stream>>>(&scalars[2], &scalars[0], &scalars[1],
                                    (float*)d_out);
}
